// ReLA_2379411882136
// MI455X (gfx1250) — compile-verified, matrix operand order checked by reading, not run on hardware
//
#include <hip/hip_runtime.h>
#include <hip/hip_bf16.h>

// ---------------------------------------------------------------------------
// ReLA transformer block for MI455X (gfx1250, wave32, WMMA bf16 path)
//   LN1 -> QKV gemm -> causal ReLU attention -> out-proj gemm -> LN2
// Matmuls: v_wmma_f32_16x16x32_bf16.  Tile movement: global_load_async_to_lds
// (ASYNCcnt).  Pipeline: wait for tile k (its DMA overlapped iteration k-1),
// barrier, issue tile k+1, compute tile k  ->  full-iteration DMA/WMMA overlap.
// ---------------------------------------------------------------------------

typedef __attribute__((ext_vector_type(16))) __bf16 v16bf;
typedef __attribute__((ext_vector_type(8)))  float  v8f;

#define WMMA_BF16(a, b, c) \
  __builtin_amdgcn_wmma_f32_16x16x32_bf16(false, (a), false, (b), (short)0, (c), false, false)

// Async 16-byte global -> LDS copy (per lane), tracked by ASYNCcnt.
__device__ __forceinline__ void async_b128(void* lds, const void* g) {
  asm volatile("global_load_async_to_lds_b128 %0, %1, off"
               :: "v"((unsigned)(size_t)lds), "v"(g) : "memory");
}
__device__ __forceinline__ void wait_async0() {
  asm volatile("s_wait_asynccnt 0" ::: "memory");
}

// Load a 16x32 bf16 A-fragment (or transposed-B fragment) from LDS.
// Per ISA 7.12.2: lane L -> row m = L&15, half h = L>>4;
// elements 0..7 = K[k0+8h .. +7], elements 8..15 = K[k0+16+8h .. +7].
__device__ __forceinline__ v16bf load_frag(const __bf16* base, int stride, int k0) {
  const int lane = threadIdx.x & 31;
  const int m = lane & 15;
  const int h = lane >> 4;
  union { v16bf v; uint4 u[2]; } r;
  const __bf16* p = base + m * stride + k0 + 8 * h;
  r.u[0] = *(const uint4*)(p);
  r.u[1] = *(const uint4*)(p + 16);
  return r.v;
}

// ---------------------------------------------------------------------------
// LayerNorm over rows of 1024; emits bf16 (GEMM A operand) or f32 (final out).
// ---------------------------------------------------------------------------
__global__ __launch_bounds__(256) void ln_kernel(const float* __restrict__ x,
                                                 const float* __restrict__ g,
                                                 const float* __restrict__ b,
                                                 __bf16* __restrict__ out_bf,
                                                 float* __restrict__ out_f) {
  const int row = blockIdx.x;
  const float* xr = x + (size_t)row * 1024;
  float s = 0.f, s2 = 0.f;
  for (int i = threadIdx.x; i < 1024; i += 256) {
    const float v = xr[i];
    s += v; s2 += v * v;
  }
  #pragma unroll
  for (int off = 16; off > 0; off >>= 1) {
    s  += __shfl_xor(s, off, 32);
    s2 += __shfl_xor(s2, off, 32);
  }
  __shared__ float ws1[8], ws2[8];
  if ((threadIdx.x & 31) == 0) { ws1[threadIdx.x >> 5] = s; ws2[threadIdx.x >> 5] = s2; }
  __syncthreads();
  s = 0.f; s2 = 0.f;
  #pragma unroll
  for (int w = 0; w < 8; ++w) { s += ws1[w]; s2 += ws2[w]; }
  const float mu   = s * (1.0f / 1024.0f);
  const float var  = s2 * (1.0f / 1024.0f) - mu * mu;
  const float rstd = rsqrtf(var + 1e-5f);
  for (int i = threadIdx.x; i < 1024; i += 256) {
    const float v = (xr[i] - mu) * rstd * g[i] + b[i];
    if (out_bf) out_bf[(size_t)row * 1024 + i] = (__bf16)v;
    if (out_f)  out_f[(size_t)row * 1024 + i]  = v;
  }
}

// ---------------------------------------------------------------------------
// One-time weight prep: Wt[n][k] = (bf16)W[k][n]   (W is K x N row-major)
// ---------------------------------------------------------------------------
__global__ __launch_bounds__(256) void transpose_cast(const float* __restrict__ W,
                                                      __bf16* __restrict__ Wt,
                                                      int K, int N) {
  __shared__ float t[32][33];
  const int n0 = blockIdx.x * 32, k0 = blockIdx.y * 32;
  const int c = threadIdx.x & 31, r4 = threadIdx.x >> 5;
  #pragma unroll
  for (int i = 0; i < 4; ++i)
    t[r4 * 4 + i][c] = W[(size_t)(k0 + r4 * 4 + i) * N + n0 + c];
  __syncthreads();
  #pragma unroll
  for (int i = 0; i < 4; ++i)
    Wt[(size_t)(n0 + r4 * 4 + i) * K + k0 + c] = (__bf16)t[c][r4 * 4 + i];
}

// ---------------------------------------------------------------------------
// Tiled bf16 GEMM:  C[4096 x N] = A[4096 x 1024] * Bt^T   (Bt is N x 1024)
// 64x128 C tile / block, 8 waves x 4 accumulators (shared A fragment), BK=32,
// double-buffered LDS fed by async copies, one barrier per k-step.
// mode 0: scatter QKV (q,k head-major [bh][n][d]; v transposed [bh][d][n]).
// mode 1: plain f32 store (out projection).
// ---------------------------------------------------------------------------
__global__ __launch_bounds__(256) void gemm64_bf16(const __bf16* __restrict__ A,
                                                   const __bf16* __restrict__ Bt,
                                                   int mode,
                                                   __bf16* __restrict__ qo,
                                                   __bf16* __restrict__ ko,
                                                   __bf16* __restrict__ vo,
                                                   float* __restrict__ yo) {
  __shared__ __bf16 As[2][64][32];    //  8 KB
  __shared__ __bf16 Bs[2][128][32];   // 16 KB
  const int bn0 = blockIdx.x * 128;
  const int bm0 = blockIdx.y * 64;
  const int tid = threadIdx.x;
  const int arow = tid >> 2, acol = (tid & 3) * 8;   // 1 chunk for A tile
  const int brow = tid >> 1, bcol = (tid & 1) * 16;  // 2 chunks for B tile
  const __bf16* aG = A  + (size_t)(bm0 + arow) * 1024 + acol;
  const __bf16* bG = Bt + (size_t)(bn0 + brow) * 1024 + bcol;

  // Prologue: tile 0 into buffer 0.
  async_b128(&As[0][arow][acol],     aG);
  async_b128(&Bs[0][brow][bcol],     bG);
  async_b128(&Bs[0][brow][bcol + 8], bG + 8);

  const int wave = tid >> 5;
  const int tr  = wave & 3;            // 16-row tile (4 x 16 = 64)
  const int tcq = (wave >> 2) * 4;     // 4 consecutive 16-col tiles (8 x 16 = 128)
  v8f acc[4] = {};

  for (int k0 = 0; k0 < 1024; k0 += 32) {
    const int cur = (k0 >> 5) & 1;
    wait_async0();      // tile k resident (DMA overlapped previous iteration)
    __syncthreads();
    if (k0 + 32 < 1024) {  // issue tile k+1; overlaps this iteration's WMMAs
      async_b128(&As[cur ^ 1][arow][acol],     aG + k0 + 32);
      async_b128(&Bs[cur ^ 1][brow][bcol],     bG + k0 + 32);
      async_b128(&Bs[cur ^ 1][brow][bcol + 8], bG + k0 + 40);
    }
    const v16bf af = load_frag(&As[cur][tr * 16][0], 32, 0);
    #pragma unroll
    for (int t = 0; t < 4; ++t) {
      const v16bf bf = load_frag(&Bs[cur][(tcq + t) * 16][0], 32, 0);
      acc[t] = WMMA_BF16(af, bf, acc[t]);
    }
  }

  // Epilogue. C/D layout: n = lane&15, m = 8*(lane>>4) + r.
  const int lane = tid & 31;
  const int n  = lane & 15;
  const int mb = (lane >> 4) * 8;
  #pragma unroll
  for (int t = 0; t < 4; ++t) {
    const int tc = tcq + t;
    #pragma unroll
    for (int r = 0; r < 8; ++r) {
      const int gr = bm0 + tr * 16 + mb + r;
      const int gc = bn0 + tc * 16 + n;
      const float val = acc[t][r];
      if (mode == 0) {
        const int which  = gc >> 10;        // 0=q 1=k 2=v
        const int within = gc & 1023;
        const int h = within >> 6, d = within & 63;
        const int bb = gr >> 11, nn = gr & 2047;
        const size_t bh = (size_t)(bb * 16 + h);
        if (which == 0)      qo[(bh * 2048 + nn) * 64 + d] = (__bf16)val;
        else if (which == 1) ko[(bh * 2048 + nn) * 64 + d] = (__bf16)val;
        else                 vo[(bh * 64 + d) * 2048 + nn] = (__bf16)val;  // transposed
      } else {
        yo[(size_t)gr * 1024 + gc] = val;
      }
    }
  }
}

// ---------------------------------------------------------------------------
// Causal ReLU attention, one (batch*head, 64-query block) per block.
//   S = scale * Q K^T ; P = relu(S) masked to j<=i ; O += P V
// K/V tiles double-buffered via async copies; only jb <= qb visited.
// Q fragments hoisted into registers across the whole j-loop.
// V arrives pre-transposed ([bh][d][n]) so the V tile is a row-wise copy.
// ---------------------------------------------------------------------------
__global__ __launch_bounds__(256) void attn_kernel(const __bf16* __restrict__ q,
                                                   const __bf16* __restrict__ k,
                                                   const __bf16* __restrict__ vt,
                                                   __bf16* __restrict__ out) {
  const int bh = blockIdx.y;   // b*16 + h
  const int qb = blockIdx.x;   // 64-row query block
  const __bf16* qp  = q  + (size_t)bh * 2048 * 64 + (size_t)qb * 64 * 64;
  const __bf16* kp  = k  + (size_t)bh * 2048 * 64;
  const __bf16* vtp = vt + (size_t)bh * 64 * 2048;

  __shared__ __bf16 Qs[64][64];      // [i][d]
  __shared__ __bf16 Ks[2][64][64];   // [j][d]
  __shared__ __bf16 Vs[2][64][64];   // [d][j]
  __shared__ __bf16 Ps[64][64];      // [i][j]

  const int tid  = threadIdx.x;
  const int crow = tid >> 2, ccol = (tid & 3) * 16;  // 32B per thread per tile

  // Prologue: Q tile + first K/V tiles (all async).
  async_b128(&Qs[crow][ccol],        qp + (size_t)crow * 64 + ccol);
  async_b128(&Qs[crow][ccol + 8],    qp + (size_t)crow * 64 + ccol + 8);
  async_b128(&Ks[0][crow][ccol],     kp + (size_t)crow * 64 + ccol);
  async_b128(&Ks[0][crow][ccol + 8], kp + (size_t)crow * 64 + ccol + 8);
  async_b128(&Vs[0][crow][ccol],     vtp + (size_t)crow * 2048 + ccol);
  async_b128(&Vs[0][crow][ccol + 8], vtp + (size_t)crow * 2048 + ccol + 8);

  const int wave = tid >> 5;
  const int tr  = wave & 3;
  const int tcb = (wave >> 2) * 2;
  const int lane = tid & 31;
  const int n  = lane & 15;
  const int mb = (lane >> 4) * 8;
  const float scale = 0.125f;  // DIM_HEAD^-0.5

  wait_async0();     // prologue tiles resident
  __syncthreads();

  // Q fragments are invariant across the whole j-loop: hoist into registers.
  const v16bf qa0 = load_frag(&Qs[tr * 16][0], 64, 0);
  const v16bf qa1 = load_frag(&Qs[tr * 16][0], 64, 32);

  v8f o0 = {}; v8f o1 = {};

  for (int jb = 0; jb <= qb; ++jb) {
    const int cur = jb & 1;
    if (jb > 0) {     // K/V tiles for jb resident (DMA overlapped iter jb-1)
      wait_async0();
      __syncthreads();
    }
    if (jb < qb) {    // issue jb+1; overlaps both WMMA phases below
      const int nj = jb + 1;
      async_b128(&Ks[cur ^ 1][crow][ccol],     kp + (size_t)(nj * 64 + crow) * 64 + ccol);
      async_b128(&Ks[cur ^ 1][crow][ccol + 8], kp + (size_t)(nj * 64 + crow) * 64 + ccol + 8);
      async_b128(&Vs[cur ^ 1][crow][ccol],     vtp + (size_t)crow * 2048 + nj * 64 + ccol);
      async_b128(&Vs[cur ^ 1][crow][ccol + 8], vtp + (size_t)crow * 2048 + nj * 64 + ccol + 8);
    }

    // S tiles (2 per wave), contraction over d in [0,64)
    #pragma unroll
    for (int t2 = 0; t2 < 2; ++t2) {
      const int tc = tcb + t2;
      v8f s = {};
      s = WMMA_BF16(qa0, load_frag(&Ks[cur][tc * 16][0], 64, 0),  s);
      s = WMMA_BF16(qa1, load_frag(&Ks[cur][tc * 16][0], 64, 32), s);
      const int gi0 = qb * 64 + tr * 16;
      const int gj  = jb * 64 + tc * 16 + n;
      #pragma unroll
      for (int r = 0; r < 8; ++r) {
        const int m = mb + r;
        float val = fmaxf(s[r] * scale, 0.0f);
        if (gj > gi0 + m) val = 0.0f;  // causal
        Ps[tr * 16 + m][tc * 16 + n] = (__bf16)val;
      }
    }
    __syncthreads();

    // O += P V, contraction over j in [0,64)
    const v16bf pa = load_frag(&Ps[tr * 16][0], 64, 0);
    const v16bf pb = load_frag(&Ps[tr * 16][0], 64, 32);
    v16bf vb0 = load_frag(&Vs[cur][tcb * 16][0], 64, 0);
    v16bf vb1 = load_frag(&Vs[cur][tcb * 16][0], 64, 32);
    o0 = WMMA_BF16(pa, vb0, o0);
    o0 = WMMA_BF16(pb, vb1, o0);
    vb0 = load_frag(&Vs[cur][(tcb + 1) * 16][0], 64, 0);
    vb1 = load_frag(&Vs[cur][(tcb + 1) * 16][0], 64, 32);
    o1 = WMMA_BF16(pa, vb0, o1);
    o1 = WMMA_BF16(pb, vb1, o1);
    // no trailing barrier needed: next iteration's wait+barrier orders reuse
  }

  // Write O (rows i, cols d) into [b, n, h*64 + d] bf16 for the out-proj GEMM.
  const int bb = bh >> 4, h = bh & 15;
  #pragma unroll
  for (int t2 = 0; t2 < 2; ++t2) {
    const v8f o = t2 ? o1 : o0;
    const int tc = tcb + t2;
    #pragma unroll
    for (int r = 0; r < 8; ++r) {
      const int i = qb * 64 + tr * 16 + mb + r;
      const int d = tc * 16 + n;
      out[((size_t)(bb * 2048 + i)) * 1024 + h * 64 + d] = (__bf16)o[r];
    }
  }
}

// ---------------------------------------------------------------------------
extern "C" void kernel_launch(void* const* d_in, const int* in_sizes, int n_in,
                              void* d_out, int out_size, void* d_ws, size_t ws_size,
                              hipStream_t stream) {
  (void)in_sizes; (void)n_in; (void)out_size; (void)ws_size;
  const float* x     = (const float*)d_in[0];
  const float* ln1_g = (const float*)d_in[1];
  const float* ln1_b = (const float*)d_in[2];
  const float* w_qkv = (const float*)d_in[3];
  const float* w_out = (const float*)d_in[4];
  const float* ln2_g = (const float*)d_in[5];
  const float* ln2_b = (const float*)d_in[6];
  float* out = (float*)d_out;

  char* ws = (char*)d_ws;
  __bf16* xn    = (__bf16*)(ws);                       //  8 MB  xn bf16 [4096,1024]
  __bf16* wqkvt = (__bf16*)(ws + (size_t)( 8u << 20)); //  6 MB  w_qkv^T bf16 [3072,1024]
  __bf16* woutt = (__bf16*)(ws + (size_t)(14u << 20)); //  2 MB  w_out^T bf16 [1024,1024]
  __bf16* qb    = (__bf16*)(ws + (size_t)(16u << 20)); //  8 MB  q  [32,2048,64]
  __bf16* kb    = (__bf16*)(ws + (size_t)(24u << 20)); //  8 MB  k  [32,2048,64]
  __bf16* vtb   = (__bf16*)(ws + (size_t)(32u << 20)); //  8 MB  v^T [32,64,2048]
  __bf16* attn  = (__bf16*)(ws + (size_t)(40u << 20)); //  8 MB  attn out bf16 [4096,1024]
  float*  y     = (float*) (ws + (size_t)(48u << 20)); // 16 MB  proj out f32 [4096,1024]

  transpose_cast<<<dim3(96, 32), 256, 0, stream>>>(w_qkv, wqkvt, 1024, 3072);
  transpose_cast<<<dim3(32, 32), 256, 0, stream>>>(w_out, woutt, 1024, 1024);
  ln_kernel<<<4096, 256, 0, stream>>>(x, ln1_g, ln1_b, xn, nullptr);
  gemm64_bf16<<<dim3(24, 64), 256, 0, stream>>>(xn, wqkvt, 0, qb, kb, vtb, nullptr);
  attn_kernel<<<dim3(32, 32), 256, 0, stream>>>(qb, kb, vtb, attn);
  gemm64_bf16<<<dim3(8, 64), 256, 0, stream>>>(attn, woutt, 1, nullptr, nullptr, nullptr, y);
  ln_kernel<<<4096, 256, 0, stream>>>(y, ln2_g, ln2_b, nullptr, out);
}
